// ReuseMultiHeadAttention_69063074120360
// MI455X (gfx1250) — matrix operand order repaired and checked by reading, not compiled, not run on hardware
//
#include <hip/hip_runtime.h>
#include <hip/hip_bf16.h>
#include <cstdint>

// ---------------------------------------------------------------------------
// CDNA5 (gfx1250) WMMA attention pipeline, bf16 inputs / f32 accumulate.
// v_wmma_f32_16x16x32_bf16:  D(16x16 f32) = A(16x32 bf16) x B(32x16 bf16) + C
// Register tiling: each wave computes a 32(M) x 64(N) strip: 2 A fragments
// are reused across 4 B fragments -> 8 WMMA per 12 b128 loads.
// ---------------------------------------------------------------------------

typedef __attribute__((ext_vector_type(16))) __bf16 v16bf;
typedef __attribute__((ext_vector_type(8)))  float  v8f;

// Problem constants
#define BB 8
#define TT 512
#define SS 512
#define DD 1024
#define NHEADS 16
#define RHEADS 8
#define NEWH 8
#define HD 64
#define PEW 1023   // 2*PE_MAX-1

__device__ __forceinline__ v8f wmma_bf16(v16bf a, v16bf b, v8f c) {
  return __builtin_amdgcn_wmma_f32_16x16x32_bf16(
      /*neg_a=*/false, a, /*neg_b=*/false, b,
      /*c_mod=*/(short)0, c, /*reuse_a=*/false, /*reuse_b=*/false);
}

// A fragment (16x32, row-major source, lda in elements).
// lane L: row = base + (L&15); K chunks at k0+8h and k0+16+8h, h = L>>4.
__device__ __forceinline__ v16bf load_a_bf16(const __bf16* A, int lda,
                                             int row, int k0, int h) {
  union { v16bf v; uint4 u[2]; } f;
  const __bf16* p = A + (size_t)row * lda + k0 + 8 * h;
  f.u[0] = *(const uint4*)(p);
  f.u[1] = *(const uint4*)(p + 16);
  return f.v;
}

// Same A fragment but from an f32 source (reuse attention scores).
__device__ __forceinline__ v16bf load_a_f32(const float* A, int lda,
                                            int row, int k0, int h) {
  const float* p0 = A + (size_t)row * lda + k0 + 8 * h;
  const float* p1 = p0 + 16;
  float4 a0 = *(const float4*)(p0);
  float4 a1 = *(const float4*)(p0 + 4);
  float4 b0 = *(const float4*)(p1);
  float4 b1 = *(const float4*)(p1 + 4);
  v16bf r;
  r[0]=(__bf16)a0.x; r[1]=(__bf16)a0.y; r[2]=(__bf16)a0.z; r[3]=(__bf16)a0.w;
  r[4]=(__bf16)a1.x; r[5]=(__bf16)a1.y; r[6]=(__bf16)a1.z; r[7]=(__bf16)a1.w;
  r[8]=(__bf16)b0.x; r[9]=(__bf16)b0.y; r[10]=(__bf16)b0.z; r[11]=(__bf16)b0.w;
  r[12]=(__bf16)b1.x; r[13]=(__bf16)b1.y; r[14]=(__bf16)b1.z; r[15]=(__bf16)b1.w;
  return r;
}

// B fragment (32x16) from Bt stored N-major (Bt[n][k], ldb in elements).
// lane L: col n = base + (L&15); K chunks at k0+8h and k0+16+8h, h = L>>4.
__device__ __forceinline__ v16bf load_b_bf16(const __bf16* Bt, int ldb,
                                             int n, int k0, int h) {
  union { v16bf v; uint4 u[2]; } f;
  const __bf16* p = Bt + (size_t)n * ldb + k0 + 8 * h;
  f.u[0] = *(const uint4*)(p);
  f.u[1] = *(const uint4*)(p + 16);
  return f.v;
}

// ---------------------------------------------------------------------------
// Prep kernels
// ---------------------------------------------------------------------------
__global__ void cvt_bf16_kernel(const float* __restrict__ src,
                                __bf16* __restrict__ dst, int n) {
  int i = blockIdx.x * blockDim.x + threadIdx.x;
  if (i < n) dst[i] = (__bf16)src[i];
}

// dst[c*R + r] = src[r*C + c]   (weight (D,N) f32 -> (N,D) bf16)
__global__ void transpose_w_kernel(const float* __restrict__ src,
                                   __bf16* __restrict__ dst, int R, int C) {
  int i = blockIdx.x * blockDim.x + threadIdx.x;
  if (i < R * C) {
    int c = i / R, r = i - c * R;
    dst[i] = (__bf16)src[(size_t)r * C + c];
  }
}

// WoT[d][k] with k = head*64+h; heads 0..7 = reuse, 8..15 = new.
__global__ void build_wot_kernel(const float* __restrict__ wo_r,
                                 const float* __restrict__ wo_n,
                                 __bf16* __restrict__ dst) {
  int i = blockIdx.x * blockDim.x + threadIdx.x;  // i = d*1024 + k
  if (i < DD * (NHEADS * HD)) {
    int d = i >> 10, k = i & 1023;
    float v = (k < RHEADS * HD) ? wo_r[(size_t)k * DD + d]
                                : wo_n[(size_t)(k - RHEADS * HD) * DD + d];
    dst[i] = (__bf16)v;
  }
}

// ---------------------------------------------------------------------------
// Generic WMMA GEMM: C[M,N] = A[M,K] * B[K,N] (+bias[n]) (*scale)
// A row-major bf16, B given as Bt[N][K] bf16. 256 threads = 8 waves laid out
// 2(m) x 4(n); block tile 64(M) x 256(N); each wave a 32x64 strip.
// EPI 0: bf16 out, (c+bias)*scale.     EPI 1: bf16 out transposed per batch:
//   out[(m>>9)*ldc*512 + n*512 + (m&511)]  (vT layout, S=512), packed b128.
// EPI 2: f32 out, c+bias.
// ---------------------------------------------------------------------------
template <int EPI>
__global__ __launch_bounds__(256) void gemm_bf16_kernel(
    const __bf16* __restrict__ A, int lda,
    const __bf16* __restrict__ Bt, int ldb,
    void* __restrict__ Cout, int ldc,
    const float* __restrict__ bias, float scale, int K) {
  const int wave = threadIdx.x >> 5;
  const int lane = threadIdx.x & 31;
  const int h = lane >> 4, ln = lane & 15;
  const int mw = wave & 1, nw = wave >> 1;
  const int m0 = blockIdx.x * 64 + mw * 32;
  const int n0 = blockIdx.y * 256 + nw * 64;

  v8f c[2][4] = {};
  for (int k0 = 0; k0 < K; k0 += 32) {
    if (k0 + 128 < K) {  // prefetch A stream ahead (global_prefetch_b8)
      __builtin_prefetch(A + (size_t)(m0 + ln) * lda + k0 + 128, 0, 3);
    }
    v16bf a0 = load_a_bf16(A, lda, m0 + ln, k0, h);
    v16bf a1 = load_a_bf16(A, lda, m0 + 16 + ln, k0, h);
#pragma unroll
    for (int e = 0; e < 4; ++e) {
      v16bf b = load_b_bf16(Bt, ldb, n0 + 16 * e + ln, k0, h);
      c[0][e] = wmma_bf16(a0, b, c[0][e]);
      c[1][e] = wmma_bf16(a1, b, c[1][e]);
    }
  }
#pragma unroll
  for (int i = 0; i < 2; ++i) {
#pragma unroll
    for (int e = 0; e < 4; ++e) {
      int n = n0 + 16 * e + ln;
      float bs = bias[n];
      if (EPI == 1) {
        // values for r=0..7 are contiguous along s in the transposed layout
        int m = m0 + 16 * i + 8 * h;  // base row of this tile half
        int b2 = m >> 9, sb = m & 511;
        union { uint4 u; __bf16 q[8]; } pk;
#pragma unroll
        for (int r = 0; r < 8; ++r) pk.q[r] = (__bf16)(c[i][e][r] + bs);
        *(uint4*)&((__bf16*)Cout)[((size_t)b2 * ldc + n) * 512 + sb] = pk.u;
      } else {
#pragma unroll
        for (int r = 0; r < 8; ++r) {
          int m = m0 + 16 * i + 8 * h + r;
          float val = c[i][e][r] + bs;
          if (EPI == 0) {
            ((__bf16*)Cout)[(size_t)m * ldc + n] = (__bf16)(val * scale);
          } else {
            ((float*)Cout)[(size_t)m * ldc + n] = val;
          }
        }
      }
    }
  }
}

// ---------------------------------------------------------------------------
// Scores + softmax for the 8 new heads.
// grid: (T/16, NEWH, B), 256 threads = 8 waves. Wave w computes the
// 16(row) x 64(col) strip at s0 = w*64 of scores = q_head * k_head^T (K=64),
// adds relative-position bias, then the block softmaxes its 16 full rows in
// LDS and writes bf16 P to workspace (packed b128 stores).
// ---------------------------------------------------------------------------
__global__ __launch_bounds__(256) void attn_scores_kernel(
    const __bf16* __restrict__ q, const __bf16* __restrict__ kk,
    const float* __restrict__ pos_emb, __bf16* __restrict__ p_out) {
  __shared__ float smem[16 * 512];
  __shared__ float pmax[16][16];
  __shared__ float psum[16][16];

  const int wave = threadIdx.x >> 5, lane = threadIdx.x & 31;
  const int h = lane >> 4, ln = lane & 15;
  const int b = blockIdx.z, head = blockIdx.y;
  const int t0 = blockIdx.x * 16;
  const int s0 = wave * 64;

  const __bf16* Ap = q + (size_t)b * TT * (NEWH * HD) + head * HD;
  const __bf16* Bp = kk + (size_t)b * SS * (NEWH * HD) + head * HD;

  v8f c[4] = {};
  for (int k0 = 0; k0 < HD; k0 += 32) {
    v16bf a = load_a_bf16(Ap, NEWH * HD, t0 + ln, k0, h);
#pragma unroll
    for (int e = 0; e < 4; ++e) {
      v16bf bb = load_b_bf16(Bp, NEWH * HD, s0 + 16 * e + ln, k0, h);
      c[e] = wmma_bf16(a, bb, c[e]);
    }
  }
  const float* pe = pos_emb + head * PEW;
#pragma unroll
  for (int e = 0; e < 4; ++e) {
    int s = s0 + 16 * e + ln;
#pragma unroll
    for (int r = 0; r < 8; ++r) {
      int t = t0 + 8 * h + r;
      int rel = s - t + 511;
      rel = rel < 0 ? 0 : (rel > 1022 ? 1022 : rel);
      smem[(8 * h + r) * 512 + s] = c[e][r] + pe[rel];
    }
  }
  __syncthreads();

  const int row = threadIdx.x >> 4;   // 0..15
  const int seg = threadIdx.x & 15;   // 16 segments of 32 cols
  float* rp = &smem[row * 512 + seg * 32];
  float mx = -1e30f;
  for (int j = 0; j < 32; ++j) mx = fmaxf(mx, rp[j]);
  pmax[row][seg] = mx;
  __syncthreads();
  float rowmax = pmax[row][0];
  for (int j = 1; j < 16; ++j) rowmax = fmaxf(rowmax, pmax[row][j]);
  float sum = 0.f;
  for (int j = 0; j < 32; ++j) {
    float ev = __expf(rp[j] - rowmax);
    rp[j] = ev;
    sum += ev;
  }
  psum[row][seg] = sum;
  __syncthreads();
  float rowsum = 0.f;
  for (int j = 0; j < 16; ++j) rowsum += psum[row][j];
  float inv = 1.f / rowsum;

  int t = t0 + row;
  __bf16* out =
      p_out + (((size_t)(b * NEWH + head) * TT + t) * SS) + seg * 32;
#pragma unroll
  for (int v4 = 0; v4 < 4; ++v4) {
    union { uint4 u; __bf16 q8[8]; } pk;
#pragma unroll
    for (int j = 0; j < 8; ++j) pk.q8[j] = (__bf16)(rp[v4 * 8 + j] * inv);
    *(uint4*)(out + v4 * 8) = pk.u;
  }
}

// ---------------------------------------------------------------------------
// P * V for all 16 heads. grid: (T/256, NHEADS, B), 8 waves each owning a
// 32(row) x 64(col) strip. Reuse heads: A = f32 reuse scores (converted to
// bf16 in-register). New heads: A = bf16 softmaxed P. B from vT (N-major).
// Output: attn_out bf16 [B*T][1024], col = head*64 + h.
// ---------------------------------------------------------------------------
__global__ __launch_bounds__(256) void attn_pv_kernel(
    const float* __restrict__ reuse_scores, const __bf16* __restrict__ p_new,
    const __bf16* __restrict__ vT, __bf16* __restrict__ attn_out) {
  const int wave = threadIdx.x >> 5, lane = threadIdx.x & 31;
  const int h = lane >> 4, ln = lane & 15;
  const int b = blockIdx.z, head = blockIdx.y;
  const int t0 = blockIdx.x * 256 + wave * 32;

  const __bf16* Bp = vT + ((size_t)b * (NHEADS * HD) + head * HD) * SS;

  v8f c[2][4] = {};
  if (head < RHEADS) {
    const float* Ap = reuse_scores + (size_t)(b * RHEADS + head) * TT * SS;
    for (int k0 = 0; k0 < SS; k0 += 32) {
      v16bf a0 = load_a_f32(Ap, SS, t0 + ln, k0, h);
      v16bf a1 = load_a_f32(Ap, SS, t0 + 16 + ln, k0, h);
#pragma unroll
      for (int e = 0; e < 4; ++e) {
        v16bf bb = load_b_bf16(Bp, SS, 16 * e + ln, k0, h);
        c[0][e] = wmma_bf16(a0, bb, c[0][e]);
        c[1][e] = wmma_bf16(a1, bb, c[1][e]);
      }
    }
  } else {
    const __bf16* Ap = p_new + (size_t)(b * NEWH + (head - RHEADS)) * TT * SS;
    for (int k0 = 0; k0 < SS; k0 += 32) {
      v16bf a0 = load_a_bf16(Ap, SS, t0 + ln, k0, h);
      v16bf a1 = load_a_bf16(Ap, SS, t0 + 16 + ln, k0, h);
#pragma unroll
      for (int e = 0; e < 4; ++e) {
        v16bf bb = load_b_bf16(Bp, SS, 16 * e + ln, k0, h);
        c[0][e] = wmma_bf16(a0, bb, c[0][e]);
        c[1][e] = wmma_bf16(a1, bb, c[1][e]);
      }
    }
  }
#pragma unroll
  for (int i = 0; i < 2; ++i) {
#pragma unroll
    for (int e = 0; e < 4; ++e) {
      int n = 16 * e + ln;
#pragma unroll
      for (int r = 0; r < 8; ++r) {
        int t = t0 + 16 * i + 8 * h + r;
        attn_out[((size_t)(b * TT + t)) * (NHEADS * HD) + head * HD + n] =
            (__bf16)c[i][e][r];
      }
    }
  }
}

// ---------------------------------------------------------------------------
// Host-side orchestration
// ---------------------------------------------------------------------------
extern "C" void kernel_launch(void* const* d_in, const int* in_sizes, int n_in,
                              void* d_out, int out_size, void* d_ws,
                              size_t ws_size, hipStream_t stream) {
  const float* query = (const float*)d_in[0];   // (B,T,D)
  const float* value = (const float*)d_in[1];   // (B,S,D)
  const float* reuse = (const float*)d_in[2];   // (B,8,T,S)
  const float* Wq    = (const float*)d_in[3];   // (D,8,64)
  const float* bq    = (const float*)d_in[4];   // (8,64)
  const float* Wk    = (const float*)d_in[5];
  const float* bk    = (const float*)d_in[6];
  const float* Wv    = (const float*)d_in[7];   // (D,16,64)
  const float* bv    = (const float*)d_in[8];   // (16,64)
  const float* pos   = (const float*)d_in[9];   // (1,8,1023)
  const float* Wo_r  = (const float*)d_in[10];  // (8,64,D)
  const float* Wo_n  = (const float*)d_in[11];  // (8,64,D)
  const float* bo    = (const float*)d_in[12];  // (D,)
  float* out = (float*)d_out;

  uint8_t* ws = (uint8_t*)d_ws;
  size_t off = 0;
  auto alloc = [&](size_t bytes) {
    void* p = ws + off;
    off += (bytes + 255) & ~(size_t)255;
    return p;
  };
  __bf16* xq   = (__bf16*)alloc((size_t)BB * TT * DD * 2);          // 8 MB
  __bf16* xv   = (__bf16*)alloc((size_t)BB * SS * DD * 2);          // 8 MB
  __bf16* wqT  = (__bf16*)alloc((size_t)NEWH * HD * DD * 2);        // 1 MB
  __bf16* wkT  = (__bf16*)alloc((size_t)NEWH * HD * DD * 2);        // 1 MB
  __bf16* wvT  = (__bf16*)alloc((size_t)NHEADS * HD * DD * 2);      // 2 MB
  __bf16* woT  = (__bf16*)alloc((size_t)DD * NHEADS * HD * 2);      // 2 MB
  __bf16* qb   = (__bf16*)alloc((size_t)BB * TT * NEWH * HD * 2);   // 4 MB
  __bf16* kb   = (__bf16*)alloc((size_t)BB * SS * NEWH * HD * 2);   // 4 MB
  __bf16* vT   = (__bf16*)alloc((size_t)BB * NHEADS * HD * SS * 2); // 8 MB
  __bf16* pnew = (__bf16*)alloc((size_t)BB * NEWH * TT * SS * 2);   // 32 MB
  __bf16* aout = (__bf16*)alloc((size_t)BB * TT * NHEADS * HD * 2); // 8 MB

  const int thr = 256;
  // --- prep: bf16 conversions + weight packing ---
  {
    int n = BB * TT * DD;
    cvt_bf16_kernel<<<(n + thr - 1) / thr, thr, 0, stream>>>(query, xq, n);
    cvt_bf16_kernel<<<(n + thr - 1) / thr, thr, 0, stream>>>(value, xv, n);
  }
  {
    int n = DD * NEWH * HD;
    transpose_w_kernel<<<(n + thr - 1) / thr, thr, 0, stream>>>(Wq, wqT, DD, NEWH * HD);
    transpose_w_kernel<<<(n + thr - 1) / thr, thr, 0, stream>>>(Wk, wkT, DD, NEWH * HD);
    int n2 = DD * NHEADS * HD;
    transpose_w_kernel<<<(n2 + thr - 1) / thr, thr, 0, stream>>>(Wv, wvT, DD, NHEADS * HD);
    build_wot_kernel<<<(n2 + thr - 1) / thr, thr, 0, stream>>>(Wo_r, Wo_n, woT);
  }
  // --- projections: q (scaled 1/8), k, vT ---
  {
    dim3 g(BB * TT / 64, (NEWH * HD) / 256);
    gemm_bf16_kernel<0><<<g, thr, 0, stream>>>(xq, DD, wqT, DD, qb, NEWH * HD,
                                               bq, 0.125f, DD);
    gemm_bf16_kernel<0><<<g, thr, 0, stream>>>(xv, DD, wkT, DD, kb, NEWH * HD,
                                               bk, 1.0f, DD);
    dim3 gv(BB * SS / 64, (NHEADS * HD) / 256);
    gemm_bf16_kernel<1><<<gv, thr, 0, stream>>>(xv, DD, wvT, DD, vT,
                                                NHEADS * HD, bv, 1.0f, DD);
  }
  // --- new-head scores + softmax ---
  {
    dim3 g(TT / 16, NEWH, BB);
    attn_scores_kernel<<<g, thr, 0, stream>>>(qb, kb, pos, pnew);
  }
  // --- P*V for all heads ---
  {
    dim3 g(TT / 256, NHEADS, BB);
    attn_pv_kernel<<<g, thr, 0, stream>>>(reuse, pnew, vT, aout);
  }
  // --- output projection ---
  {
    dim3 g(BB * TT / 64, DD / 256);
    gemm_bf16_kernel<2><<<g, thr, 0, stream>>>(aout, NHEADS * HD, woT, DD, out,
                                               DD, bo, 1.0f, DD);
  }
  (void)in_sizes; (void)n_in; (void)out_size; (void)ws_size;
}
